// LSTM_27144193311539
// MI455X (gfx1250) — matrix operand order repaired and checked by reading, not compiled, not run on hardware
//
#include <hip/hip_runtime.h>
#include <hip/hip_bf16.h>

// ---------------------------------------------------------------------------
// Seq2seq LSTM (enc 2-layer + dec 2-layer + FC head) for MI455X / gfx1250.
// Persistent wave32 kernel, bf16 WMMA (v_wmma_f32_16x16x32_bf16), fp32 cell
// state, weights repacked into WMMA fragment layout (L2-resident, ~52MB).
// ---------------------------------------------------------------------------

#define EPSBN 1e-5f

constexpr int Bsz = 64;    // batch
constexpr int Tt  = 1024;  // time steps
constexpr int Dd  = 128;   // input dim
constexpr int Hh  = 1024;  // hidden
constexpr int Cc  = 128;   // classes
constexpr int NWG = 64;    // persistent workgroups (one per 16 hidden channels)
constexpr int BLK = 256;   // threads per WG -> 8 waves (wave32)

typedef __attribute__((ext_vector_type(16))) __bf16 v16bf;
typedef __attribute__((ext_vector_type(8)))  float  v8f;

struct Params {
  // raw fp32 inputs
  const float *x;
  const float *ibn_g, *ibn_b, *ibn_m, *ibn_v;
  const float *enc_b0, *enc_b1, *dec_b0, *dec_b1;
  const float *hbn_g, *hbn_b, *hbn_m, *hbn_v;
  const float *fc1_b;
  const float *fbn_g, *fbn_b, *fbn_m, *fbn_v;
  const float *fc2_b;
  // fragment-packed bf16 weights (in workspace)
  const __bf16 *WencIh0, *WencHh0, *WencIh1, *WencHh1;
  const __bf16 *WdecIh0, *WdecHh0, *WdecIh1, *WdecHh1;
  const __bf16 *Wfc1, *Wfc2;
  // state (workspace)
  __bf16 *h0f0, *h0f1;     // layer0 hidden, fragment layout, double buffered
  __bf16 *h1f0, *h1f1;     // layer1 hidden
  __bf16 *hbfrag;          // BN(d1) fragments (fc1 input)
  __bf16 *f1frag;          // relu(BN(fc1)) fragments (fc2 input)
  __bf16 *xfrag;           // BN(x_t) fragments (64x128)
  __bf16 *yfrag;           // decoder feedback pred fragments (64x128)
  float  *c0, *c1;         // cell states fp32 (64x1024)
  unsigned *bar;           // [0]=arrive counter, [1]=generation
  float  *out;             // (B,T,C) fp32
};

// ----------------------------- device helpers ------------------------------

__device__ inline v8f wmma_bf16(v16bf a, v16bf b, v8f c) {
  return __builtin_amdgcn_wmma_f32_16x16x32_bf16(false, a, false, b,
                                                 (short)0, c, false, false);
}

__device__ inline float sigm(float x) { return 1.0f / (1.0f + __expf(-x)); }

// A-matrix 16-bit fragment k mapping (16x32 tile):
//   k = (j/8)*16 + half*8 + (j%8),  half = lane/16, row m = lane%16
// Scatter one (batch b, channel ch) bf16 value into A-fragment layout.
__device__ inline void scatter_afrag(__bf16* dst, int nkt, int b, int ch, float v) {
  int kt   = ch >> 5;
  int kl   = ch & 31;
  int j    = ((kl >> 4) << 3) + (kl & 7);
  int half = (kl >> 3) & 1;
  int ln   = half * 16 + (b & 15);
  dst[(((size_t)((b >> 4) * nkt + kt) * 32 + ln) << 4) + j] = (__bf16)v;
}

// grid-wide sense barrier (agent scope atomics + s_sleep spin)
__device__ inline void gsync(unsigned* bar, unsigned& gen) {
  __syncthreads();
  ++gen;
  if (threadIdx.x == 0) {
    unsigned prev = __hip_atomic_fetch_add(&bar[0], 1u, __ATOMIC_ACQ_REL,
                                           __HIP_MEMORY_SCOPE_AGENT);
    if (prev == (unsigned)NWG - 1u) {
      __hip_atomic_store(&bar[0], 0u, __ATOMIC_RELAXED, __HIP_MEMORY_SCOPE_AGENT);
      __hip_atomic_fetch_add(&bar[1], 1u, __ATOMIC_RELEASE, __HIP_MEMORY_SCOPE_AGENT);
    } else {
      while (__hip_atomic_load(&bar[1], __ATOMIC_ACQUIRE,
                               __HIP_MEMORY_SCOPE_AGENT) < gen)
        __builtin_amdgcn_s_sleep(2);
    }
  }
  __syncthreads();
}

// Build BN(x[:,t,:]) as A-fragments (64x128 -> 16 fragments, 8192 bf16 elems).
__device__ inline void build_xfrag(const Params& P, int t) {
  int gid = blockIdx.x * BLK + threadIdx.x;
  if (gid < Bsz * Dd) {
    int j    = gid & 15;
    int lane = (gid >> 4) & 31;
    int kt   = (gid >> 9) & 3;
    int mt   = gid >> 11;
    int m    = mt * 16 + (lane & 15);
    int k    = kt * 32 + ((j >> 3) << 4) + ((lane >> 4) << 3) + (j & 7);
    float xv = P.x[((size_t)m * Tt + t) * Dd + k];
    float sc = P.ibn_g[k] * rsqrtf(P.ibn_v[k] + EPSBN);
    float v  = (xv - P.ibn_m[k]) * sc + P.ibn_b[k];
    P.xfrag[gid] = (__bf16)v;
  }
}

// One LSTM layer for one step: z = A0@W0^T + A1@W1^T + bias, then gates.
// WG w owns hidden channels [16w,16w+16) across all 4 gates (i,f,g,o).
// Each wave: mt = wave&3, gates {2*(wave>>2), 2*(wave>>2)+1} -> 2 tiles.
__device__ void lstm_phase(const Params& P,
                           const __bf16* A0, const __bf16* W0, int nkt0,
                           const __bf16* A1, const __bf16* W1, int nkt1,
                           const float* bias, float* cst, __bf16* hout,
                           const float* bng, const float* bnb,
                           const float* bnm, const float* bnv, __bf16* hbout,
                           float* lds) {
  const int w    = blockIdx.x;
  const int tid  = threadIdx.x;
  const int lane = tid & 31;
  const int wv   = tid >> 5;
  const int mt   = wv & 3;
  const int gp   = wv >> 2;
  const int g0   = gp * 2, g1 = gp * 2 + 1;
  const int nt0  = g0 * (Hh / 16) / 1 * 1;  // placeholder, fixed below
  (void)nt0;
  const int ntA  = g0 * 64 + w;             // gate-g0 tile column
  const int ntB  = g1 * 64 + w;             // gate-g1 tile column

  v8f acc0 = {0.f,0.f,0.f,0.f,0.f,0.f,0.f,0.f};
  v8f acc1 = {0.f,0.f,0.f,0.f,0.f,0.f,0.f,0.f};

  { // input contribution
    const __bf16* ab = A0 + ((size_t)(mt * nkt0) << 9) + lane * 16;
    const __bf16* b0 = W0 + ((size_t)(ntA * nkt0) << 9) + lane * 16;
    const __bf16* b1 = W0 + ((size_t)(ntB * nkt0) << 9) + lane * 16;
    for (int kf = 0; kf < nkt0; ++kf) {
      v16bf a  = *(const v16bf*)(ab + (kf << 9));
      v16bf w0 = *(const v16bf*)(b0 + (kf << 9));
      v16bf w1 = *(const v16bf*)(b1 + (kf << 9));
      acc0 = wmma_bf16(a, w0, acc0);
      acc1 = wmma_bf16(a, w1, acc1);
    }
  }
  { // hidden contribution
    const __bf16* ab = A1 + ((size_t)(mt * nkt1) << 9) + lane * 16;
    const __bf16* b0 = W1 + ((size_t)(ntA * nkt1) << 9) + lane * 16;
    const __bf16* b1 = W1 + ((size_t)(ntB * nkt1) << 9) + lane * 16;
    for (int kf = 0; kf < nkt1; ++kf) {
      v16bf a  = *(const v16bf*)(ab + (kf << 9));
      v16bf w0 = *(const v16bf*)(b0 + (kf << 9));
      v16bf w1 = *(const v16bf*)(b1 + (kf << 9));
      acc0 = wmma_bf16(a, w0, acc0);
      acc1 = wmma_bf16(a, w1, acc1);
    }
  }

  // D layout: VGPR r, lane l -> row = r + (l/16)*8, col = l%16
  const int col = lane & 15, rb = (lane >> 4) * 8;
  for (int r = 0; r < 8; ++r) {
    int row = rb + r;
    lds[((g0 * 4 + mt) * 16 + row) * 16 + col] = acc0[r];
    lds[((g1 * 4 + mt) * 16 + row) * 16 + col] = acc1[r];
  }
  __syncthreads();

  // gate nonlinearity over the WG's 64x16 slice
  for (int p = tid; p < Bsz * 16; p += BLK) {
    int b   = p >> 4;
    int hl  = p & 15;
    int hg  = w * 16 + hl;
    int mtb = b >> 4, rr = b & 15;
    float zi = lds[((0 * 4 + mtb) * 16 + rr) * 16 + hl] + bias[hg];
    float zf = lds[((1 * 4 + mtb) * 16 + rr) * 16 + hl] + bias[Hh + hg];
    float zg = lds[((2 * 4 + mtb) * 16 + rr) * 16 + hl] + bias[2 * Hh + hg];
    float zo = lds[((3 * 4 + mtb) * 16 + rr) * 16 + hl] + bias[3 * Hh + hg];
    float cc = cst[(size_t)b * Hh + hg];
    float cn = sigm(zf) * cc + sigm(zi) * tanhf(zg);
    cst[(size_t)b * Hh + hg] = cn;
    float h = sigm(zo) * tanhf(cn);
    scatter_afrag(hout, Hh / 32, b, hg, h);
    if (hbout) {
      float sc = bng[hg] * rsqrtf(bnv[hg] + EPSBN);
      float hb = (h - bnm[hg]) * sc + bnb[hg];
      scatter_afrag(hbout, Hh / 32, b, hg, hb);
    }
  }
  __syncthreads();
}

// fc1: f1 = relu(BN(hb @ fc1_W^T + fc1_b)). WG w owns N-tile w (16 channels),
// 4 M-tiles; wave pairs split K (kf 0..15 / 16..31), reduce in LDS.
__device__ void fc1_phase(const Params& P, float* lds) {
  const int w    = blockIdx.x;
  const int tid  = threadIdx.x;
  const int lane = tid & 31;
  const int wv   = tid >> 5;
  const int mt   = wv >> 1;
  const int kh   = wv & 1;

  v8f acc = {0.f,0.f,0.f,0.f,0.f,0.f,0.f,0.f};
  const __bf16* ab = P.hbfrag + ((size_t)(mt * 32 + kh * 16) << 9) + lane * 16;
  const __bf16* bb = P.Wfc1   + ((size_t)(w  * 32 + kh * 16) << 9) + lane * 16;
  for (int kf = 0; kf < 16; ++kf) {
    v16bf a  = *(const v16bf*)(ab + (kf << 9));
    v16bf wB = *(const v16bf*)(bb + (kf << 9));
    acc = wmma_bf16(a, wB, acc);
  }
  const int col = lane & 15, rb = (lane >> 4) * 8;
  for (int r = 0; r < 8; ++r)
    lds[wv * 256 + (rb + r) * 16 + col] = acc[r];
  __syncthreads();

  for (int p = tid; p < Bsz * 16; p += BLK) {
    int b   = p >> 4;
    int hl  = p & 15;
    int hg  = w * 16 + hl;
    int mtb = b >> 4, rr = b & 15;
    float v = lds[(mtb * 2 + 0) * 256 + rr * 16 + hl] +
              lds[(mtb * 2 + 1) * 256 + rr * 16 + hl] + P.fc1_b[hg];
    float sc = P.fbn_g[hg] * rsqrtf(P.fbn_v[hg] + EPSBN);
    v = fmaxf((v - P.fbn_m[hg]) * sc + P.fbn_b[hg], 0.0f);
    scatter_afrag(P.f1frag, Hh / 32, b, hg, v);
  }
  __syncthreads();
}

// fc2: pred = f1 @ fc2_W^T + fc2_b. 32 output tiles -> WGs 0..31 one tile,
// 8 waves split K (4 kfrags each), reduce in LDS, write out + yfrag feedback.
__device__ void fc2_phase(const Params& P, int t, float* lds) {
  const int w = blockIdx.x;
  if (w < 32) {
    const int tid  = threadIdx.x;
    const int lane = tid & 31;
    const int wv   = tid >> 5;
    const int mt   = w >> 3;
    const int nt   = w & 7;

    v8f acc = {0.f,0.f,0.f,0.f,0.f,0.f,0.f,0.f};
    const __bf16* ab = P.f1frag + ((size_t)(mt * 32 + wv * 4) << 9) + lane * 16;
    const __bf16* bb = P.Wfc2   + ((size_t)(nt * 32 + wv * 4) << 9) + lane * 16;
    for (int kf = 0; kf < 4; ++kf) {
      v16bf a  = *(const v16bf*)(ab + (kf << 9));
      v16bf wB = *(const v16bf*)(bb + (kf << 9));
      acc = wmma_bf16(a, wB, acc);
    }
    const int col = lane & 15, rb = (lane >> 4) * 8;
    for (int r = 0; r < 8; ++r)
      lds[wv * 256 + (rb + r) * 16 + col] = acc[r];
    __syncthreads();

    // 256 threads, one element each
    int row = tid >> 4, c2 = tid & 15;
    float v = 0.f;
    for (int q = 0; q < 8; ++q) v += lds[q * 256 + tid];
    int b = mt * 16 + row;
    int c = nt * 16 + c2;
    v += P.fc2_b[c];
    P.out[((size_t)b * Tt + t) * Cc + c] = v;
    scatter_afrag(P.yfrag, Cc / 32, b, c, v);
    __syncthreads();
  }
}

// ------------------------------ kernels ------------------------------------

// Repack fp32 weight (N x K row-major) into WMMA B-fragment layout:
// dst[nt][kt][lane][j], B tile 32x16: col n = lane%16, k = (j/8)*16 + (lane/16)*8 + j%8.
__global__ __launch_bounds__(BLK) void pack_w(const float* __restrict__ W,
                                              __bf16* __restrict__ dst,
                                              int N, int K) {
  int idx = blockIdx.x * BLK + threadIdx.x;
  int total = N * K;
  if (idx >= total) return;
  int j    = idx & 15;
  int lane = (idx >> 4) & 31;
  int nkt  = K >> 5;
  int kt   = (idx >> 9) % nkt;
  int nt   = idx / (nkt << 9);
  int n    = nt * 16 + (lane & 15);
  int k    = kt * 32 + ((j >> 3) << 4) + ((lane >> 4) << 3) + (j & 7);
  dst[idx] = (__bf16)W[(size_t)n * K + k];
}

__global__ __launch_bounds__(BLK) void lstm_persist(Params P) {
  __shared__ float lds[4096];
  unsigned gen = 0;

  build_xfrag(P, 0);
  gsync(P.bar, gen);

  // -------- encoder --------
  for (int t = 0; t < Tt; ++t) {
    const int par = t & 1;
    __bf16* h0r = par ? P.h0f1 : P.h0f0;
    __bf16* h0w = par ? P.h0f0 : P.h0f1;
    __bf16* h1r = par ? P.h1f1 : P.h1f0;
    __bf16* h1w = par ? P.h1f0 : P.h1f1;

    lstm_phase(P, P.xfrag, P.WencIh0, Dd / 32,
               h0r, P.WencHh0, Hh / 32,
               P.enc_b0, P.c0, h0w,
               nullptr, nullptr, nullptr, nullptr, nullptr, lds);
    gsync(P.bar, gen);

    lstm_phase(P, h0w, P.WencIh1, Hh / 32,
               h1r, P.WencHh1, Hh / 32,
               P.enc_b1, P.c1, h1w,
               nullptr, nullptr, nullptr, nullptr, nullptr, lds);
    if (t + 1 < Tt) build_xfrag(P, t + 1);
    gsync(P.bar, gen);
  }

  // -------- decoder (states continue from encoder; y0 = 0 via memset) -----
  for (int t = 0; t < Tt; ++t) {
    const int par = t & 1;
    __bf16* h0r = par ? P.h0f1 : P.h0f0;
    __bf16* h0w = par ? P.h0f0 : P.h0f1;
    __bf16* h1r = par ? P.h1f1 : P.h1f0;
    __bf16* h1w = par ? P.h1f0 : P.h1f1;

    lstm_phase(P, P.yfrag, P.WdecIh0, Cc / 32,
               h0r, P.WdecHh0, Hh / 32,
               P.dec_b0, P.c0, h0w,
               nullptr, nullptr, nullptr, nullptr, nullptr, lds);
    gsync(P.bar, gen);

    lstm_phase(P, h0w, P.WdecIh1, Hh / 32,
               h1r, P.WdecHh1, Hh / 32,
               P.dec_b1, P.c1, h1w,
               P.hbn_g, P.hbn_b, P.hbn_m, P.hbn_v, P.hbfrag, lds);
    gsync(P.bar, gen);

    fc1_phase(P, lds);
    gsync(P.bar, gen);

    fc2_phase(P, t, lds);
    gsync(P.bar, gen);
  }
}

// ------------------------------- host side ---------------------------------

extern "C" void kernel_launch(void* const* d_in, const int* in_sizes, int n_in,
                              void* d_out, int out_size, void* d_ws, size_t ws_size,
                              hipStream_t stream) {
  (void)in_sizes; (void)n_in; (void)out_size;

  char* base = (char*)d_ws;
  size_t off = 0;
  auto alloc = [&](size_t bytes) -> char* {
    off = (off + 255) & ~(size_t)255;
    char* p = base + off;
    off += bytes;
    return p;
  };

  // packed bf16 weights
  __bf16* WencIh0 = (__bf16*)alloc((size_t)4 * Hh * Dd * 2);
  __bf16* WencHh0 = (__bf16*)alloc((size_t)4 * Hh * Hh * 2);
  __bf16* WencIh1 = (__bf16*)alloc((size_t)4 * Hh * Hh * 2);
  __bf16* WencHh1 = (__bf16*)alloc((size_t)4 * Hh * Hh * 2);
  __bf16* WdecIh0 = (__bf16*)alloc((size_t)4 * Hh * Cc * 2);
  __bf16* WdecHh0 = (__bf16*)alloc((size_t)4 * Hh * Hh * 2);
  __bf16* WdecIh1 = (__bf16*)alloc((size_t)4 * Hh * Hh * 2);
  __bf16* WdecHh1 = (__bf16*)alloc((size_t)4 * Hh * Hh * 2);
  __bf16* Wfc1    = (__bf16*)alloc((size_t)Hh * Hh * 2);
  __bf16* Wfc2    = (__bf16*)alloc((size_t)Cc * Hh * 2);

  // state region (zeroed every call)
  off = (off + 255) & ~(size_t)255;
  size_t state_off = off;
  __bf16* h0f0   = (__bf16*)alloc((size_t)Bsz * Hh * 2);
  __bf16* h0f1   = (__bf16*)alloc((size_t)Bsz * Hh * 2);
  __bf16* h1f0   = (__bf16*)alloc((size_t)Bsz * Hh * 2);
  __bf16* h1f1   = (__bf16*)alloc((size_t)Bsz * Hh * 2);
  __bf16* hbfrag = (__bf16*)alloc((size_t)Bsz * Hh * 2);
  __bf16* f1frag = (__bf16*)alloc((size_t)Bsz * Hh * 2);
  __bf16* xfrag  = (__bf16*)alloc((size_t)Bsz * Dd * 2);
  __bf16* yfrag  = (__bf16*)alloc((size_t)Bsz * Cc * 2);
  float*  c0     = (float*)alloc((size_t)Bsz * Hh * 4);
  float*  c1     = (float*)alloc((size_t)Bsz * Hh * 4);
  unsigned* bar  = (unsigned*)alloc(256);
  size_t state_bytes = off - state_off;

  if (off > ws_size) return;  // workspace too small; nothing safe to do

  hipMemsetAsync(base + state_off, 0, state_bytes, stream);

  auto pack = [&](int src_idx, __bf16* dst, int N, int K) {
    int total = N * K;
    pack_w<<<dim3((total + BLK - 1) / BLK), dim3(BLK), 0, stream>>>(
        (const float*)d_in[src_idx], dst, N, K);
  };
  pack(5,  WencIh0, 4 * Hh, Dd);
  pack(6,  WencHh0, 4 * Hh, Hh);
  pack(8,  WencIh1, 4 * Hh, Hh);
  pack(9,  WencHh1, 4 * Hh, Hh);
  pack(11, WdecIh0, 4 * Hh, Cc);
  pack(12, WdecHh0, 4 * Hh, Hh);
  pack(14, WdecIh1, 4 * Hh, Hh);
  pack(15, WdecHh1, 4 * Hh, Hh);
  pack(21, Wfc1,    Hh, Hh);
  pack(27, Wfc2,    Cc, Hh);

  Params P;
  P.x = (const float*)d_in[0];
  P.ibn_g = (const float*)d_in[1];  P.ibn_b = (const float*)d_in[2];
  P.ibn_m = (const float*)d_in[3];  P.ibn_v = (const float*)d_in[4];
  P.enc_b0 = (const float*)d_in[7];  P.enc_b1 = (const float*)d_in[10];
  P.dec_b0 = (const float*)d_in[13]; P.dec_b1 = (const float*)d_in[16];
  P.hbn_g = (const float*)d_in[17]; P.hbn_b = (const float*)d_in[18];
  P.hbn_m = (const float*)d_in[19]; P.hbn_v = (const float*)d_in[20];
  P.fc1_b = (const float*)d_in[22];
  P.fbn_g = (const float*)d_in[23]; P.fbn_b = (const float*)d_in[24];
  P.fbn_m = (const float*)d_in[25]; P.fbn_v = (const float*)d_in[26];
  P.fc2_b = (const float*)d_in[28];
  P.WencIh0 = WencIh0; P.WencHh0 = WencHh0;
  P.WencIh1 = WencIh1; P.WencHh1 = WencHh1;
  P.WdecIh0 = WdecIh0; P.WdecHh0 = WdecHh0;
  P.WdecIh1 = WdecIh1; P.WdecHh1 = WdecHh1;
  P.Wfc1 = Wfc1; P.Wfc2 = Wfc2;
  P.h0f0 = h0f0; P.h0f1 = h0f1; P.h1f0 = h1f0; P.h1f1 = h1f1;
  P.hbfrag = hbfrag; P.f1frag = f1frag; P.xfrag = xfrag; P.yfrag = yfrag;
  P.c0 = c0; P.c1 = c1; P.bar = bar;
  P.out = (float*)d_out;

  lstm_persist<<<dim3(NWG), dim3(BLK), 0, stream>>>(P);
}
